// LieModel_SO2_90958817394731
// MI455X (gfx1250) — matrix operand order repaired and checked by reading, not compiled, not run on hardware
//
#include <hip/hip_runtime.h>
#include <cstddef>

// MI455X / gfx1250, wave32. bf16 WMMA (f32 accum) for all matmuls.
//
// Pipeline (all deterministic, no atomics):
//   prep_x    : x[256,784] f32 -> xb[256,800] bf16 (K padded to 800)
//   prep_small: params->pb[32,64] bf16, W_lin->wlb[64,16] bf16 (transposed),
//               zb[..,p=31] = 0
//   gemm1     : Z = xb . T^T (M=256, N'=24304, K=784->800); double-buffered LDS,
//               A panel staged with GLOBAL_LOAD_ASYNC_TO_LDS_B128 (ASYNCcnt),
//               B panel (T) f32->bf16 packed-pair transposed store (ds_store_b32).
//   stage2    : Y = Z.pb (4 WMMA), LDS relayout, U = Y.wlb (2 WMMA),
//               +b_lin, swish, shfl-max over 16 lanes -> s[256*784] f32
//   head      : out[256,10] = s @ W1^T + b1 (deterministic block reduction)

typedef __bf16 bf16;
typedef __bf16 v16bf __attribute__((ext_vector_type(16)));
typedef __bf16 v8bf  __attribute__((ext_vector_type(8)));
typedef float  v8f   __attribute__((ext_vector_type(8)));
typedef int    v4i   __attribute__((vector_size(16)));

#if defined(__has_builtin)
#if __has_builtin(__builtin_amdgcn_global_load_async_to_lds_b128)
#define ASYNC_LDS 1
#endif
#endif

__device__ __forceinline__ v16bf cat16(v8bf lo, v8bf hi) {
  return __builtin_shufflevector(lo, hi, 0,1,2,3,4,5,6,7,8,9,10,11,12,13,14,15);
}
__device__ __forceinline__ v8bf ld8(const bf16* p) { return *(const v8bf*)p; }

__device__ __forceinline__ v8f wmma_bf16(v16bf a, v16bf b, v8f c) {
  return __builtin_amdgcn_wmma_f32_16x16x32_bf16(false, a, false, b,
                                                 (short)0, c, false, false);
}

// pack two f32 -> packed bf16 pair (lo in [15:0], hi in [31:16])
__device__ __forceinline__ unsigned pack2(float lo, float hi) {
  unsigned short a = __builtin_bit_cast(unsigned short, (bf16)lo);
  unsigned short b = __builtin_bit_cast(unsigned short, (bf16)hi);
  return ((unsigned)b << 16) | a;
}

#ifdef ASYNC_LDS
typedef __attribute__((address_space(1))) v4i gv4i;   // global int4
typedef __attribute__((address_space(3))) v4i lv4i;   // LDS int4
__device__ __forceinline__ void async_copy16(const bf16* g, bf16* l) {
  __builtin_amdgcn_global_load_async_to_lds_b128(
      (gv4i*)(void*)g, (lv4i*)(void*)l, /*offset=*/0, /*cpol=*/0);
}
__device__ __forceinline__ void wait_async0() {
#if __has_builtin(__builtin_amdgcn_s_wait_asynccnt)
  __builtin_amdgcn_s_wait_asynccnt(0);
#else
  asm volatile("s_wait_asynccnt 0" ::: "memory");
#endif
}
#endif

// ---------------------------------------------------------------- prep kernels
__global__ __launch_bounds__(256) void prep_x(const float* __restrict__ x,
                                              bf16* __restrict__ xb) {
  int t = blockIdx.x * 256 + threadIdx.x;          // 256*800 = 204800 exact
  int b = t / 800, k = t - b * 800;
  xb[t] = (k < 784) ? (bf16)x[b * 784 + k] : (bf16)0.f;
}

__global__ __launch_bounds__(256) void prep_small(const float* __restrict__ params,
                                                  const float* __restrict__ W_lin,
                                                  bf16* __restrict__ pb,
                                                  bf16* __restrict__ wlb,
                                                  bf16* __restrict__ zb) {
  int t = blockIdx.x * 256 + threadIdx.x;          // 200704 threads
  if (t < 200704) zb[(size_t)t * 32 + 31] = (bf16)0.f;      // pad p=31 lane
  if (t < 2048)  pb[t]  = (t < 31 * 64) ? (bf16)params[t] : (bf16)0.f;
  if (t < 1024) { int d = t >> 4, e = t & 15; wlb[t] = (bf16)W_lin[e * 64 + d]; }
}

// ---------------------------------------------------------------- main GEMM
// One WG per 112-wide column tile covering the full M=256 batch, so every
// element of T is read from HBM exactly once (76 MB -> ~3.3 us at 23.3 TB/s).
// Software-pipelined: while buffer `cur` feeds the 14 WMMAs, buffer `nxt` is
// filled (A via async-DMA to LDS, B via regs loaded before the WMMA chain).
// B staging: 448 chunks, each = 2 n-rows x 4 k; statically-indexed two phases
// (phase 0: chunk t for all threads, phase 1: chunk t+256 for t<192).
__global__ __launch_bounds__(256) void gemm1(const bf16* __restrict__ xb,
                                             const float* __restrict__ T,
                                             bf16* __restrict__ zb) {
  constexpr int LDA = 40;   // 32 + 8 pad (bf16 elems)
  constexpr int LDB = 120;  // 112 + 8 pad
  __shared__ __align__(16) bf16 As[2][256 * LDA];   // 2 x 20.0 KB
  __shared__ __align__(16) bf16 Bs[2][32 * LDB];    // 2 x  7.5 KB
  const int t = threadIdx.x;
  const int lane = t & 31, wave = t >> 5;
  const int nbase = blockIdx.x * 112;               // 217 tiles: 31 p-panels * 7
  const int half = lane >> 4, rr = lane & 15, kh = half * 8;

  v8f c[2][7] = {};                                 // wave: 32 rows x 112 cols

#ifndef ASYNC_LDS
  v8bf areg[4];
#endif
  float4 t0a, t0b, t1a, t1b;                        // B staging registers

  // ---- staging helpers -----------------------------------------------------
  auto stageA_issue = [&](int k0, int buf) {
#pragma unroll
    for (int it = 0; it < 4; ++it) {                // 1024 chunks / 256 thr
      int ci = t + it * 256;
      int row = ci >> 2, kc = (ci & 3) * 8;
#ifdef ASYNC_LDS
      async_copy16(&xb[(size_t)row * 800 + k0 + kc], &As[buf][row * LDA + kc]);
#else
      areg[it] = ld8(&xb[(size_t)row * 800 + k0 + kc]);
#endif
    }
  };
  auto stageA_commit = [&](int buf) {
#ifdef ASYNC_LDS
    wait_async0();
#else
#pragma unroll
    for (int it = 0; it < 4; ++it) {
      int ci = t + it * 256;
      int row = ci >> 2, kc = (ci & 3) * 8;
      *(v8bf*)&As[buf][row * LDA + kc] = areg[it];
    }
#endif
  };
  auto loadT = [&](int k0, int ci, float4& ra, float4& rb) {
    int np = ci >> 3, kc = (ci & 7) * 4;            // row-pair, k-offset
    int k = k0 + kc;
    if (k + 3 < 784) {                              // 784 % 4 == 0: all-or-nothing
      ra = *(const float4*)&T[(size_t)(nbase + 2 * np)     * 784 + k];
      rb = *(const float4*)&T[(size_t)(nbase + 2 * np + 1) * 784 + k];
      if (k + 64 < 784) {                           // global_prefetch_b8, 2 ahead
        __builtin_prefetch(&T[(size_t)(nbase + 2 * np)     * 784 + k + 64], 0, 0);
        __builtin_prefetch(&T[(size_t)(nbase + 2 * np + 1) * 784 + k + 64], 0, 0);
      }
    } else {
      ra = make_float4(0.f, 0.f, 0.f, 0.f);
      rb = make_float4(0.f, 0.f, 0.f, 0.f);
    }
  };
  auto storeT = [&](int buf, int ci, const float4& ra, const float4& rb) {
    int np = ci >> 3, kc = (ci & 7) * 4;
    const float a[4] = {ra.x, ra.y, ra.z, ra.w};
    const float b[4] = {rb.x, rb.y, rb.z, rb.w};
#pragma unroll
    for (int j = 0; j < 4; ++j)                     // packed-pair ds_store_b32
      *(unsigned*)&Bs[buf][(kc + j) * LDB + 2 * np] = pack2(a[j], b[j]);
  };
  auto stageB_load = [&](int k0) {
    loadT(k0, t, t0a, t0b);
    if (t < 192) loadT(k0, t + 256, t1a, t1b);
  };
  auto stageB_store = [&](int buf) {
    storeT(buf, t, t0a, t0b);
    if (t < 192) storeT(buf, t + 256, t1a, t1b);
  };

  // ---- prologue: fill buffer 0 --------------------------------------------
  stageA_issue(0, 0);
  stageB_load(0);
  stageB_store(0);
  stageA_commit(0);
  __syncthreads();

  // ---- pipelined K loop: 25 steps of K=32 ---------------------------------
  for (int i = 0; i < 25; ++i) {
    const int cur = i & 1, nxt = cur ^ 1;
    if (i < 24) {                                   // start filling nxt
      stageA_issue((i + 1) * 32, nxt);
      stageB_load((i + 1) * 32);
    }

    // A frags: lane<16 -> row M=lane, K {0..7,16..23}; lane>=16 -> K {8..15,24..31}
    const bf16* a0p = &As[cur][(wave * 32 + rr) * LDA];
    const bf16* a1p = &As[cur][(wave * 32 + 16 + rr) * LDA];
    v16bf a0 = cat16(ld8(&a0p[kh]), ld8(&a0p[kh + 16]));
    v16bf a1 = cat16(ld8(&a1p[kh]), ld8(&a1p[kh + 16]));
    // B frags: lane l holds column N=l%16, K {0..7,16..23} (lane<16) / {8..15,24..31}
    v16bf bf[7];
#pragma unroll
    for (int j = 0; j < 7; ++j) {
#pragma unroll
      for (int s = 0; s < 16; ++s)
        bf[j][s] = Bs[cur][(kh + (s & 7) + (s >> 3) * 16) * LDB + j * 16 + rr];
    }
#pragma unroll
    for (int j = 0; j < 7; ++j) {
      c[0][j] = wmma_bf16(a0, bf[j], c[0][j]);
      c[1][j] = wmma_bf16(a1, bf[j], c[1][j]);
    }

    if (i < 24) {                                   // publish nxt
      stageB_store(nxt);
      stageA_commit(nxt);
    }
    __syncthreads();
  }

  // ---- epilogue: scatter C (f32) as bf16 into zb[(b*784+m)*32 + p] --------
#pragma unroll
  for (int j = 0; j < 7; ++j) {
    int np = nbase + j * 16 + rr;                   // n' = p*784 + m
    int p  = np / 784;
    int m  = np - p * 784;
#pragma unroll
    for (int g = 0; g < 2; ++g)
#pragma unroll
      for (int v = 0; v < 8; ++v) {
        int b = wave * 32 + g * 16 + v + 8 * half;  // C: M = v + 8*(lane>>4)
        zb[((size_t)b * 784 + m) * 32 + p] = (bf16)c[g][j][v];
      }
  }
}

// ------------------------------------------------- fused epilogue GEMMs + swish/max
__global__ __launch_bounds__(256) void stage2(const bf16* __restrict__ zb,
                                              const bf16* __restrict__ pb,
                                              const bf16* __restrict__ wlb,
                                              const float* __restrict__ b_lin,
                                              float* __restrict__ sout) {
  constexpr int LDY = 72;                        // 64 + 8 pad
  __shared__ __align__(16) bf16 Ys[8][16 * LDY]; // per-wave Y tile, 18 KB total
  const int t = threadIdx.x, lane = t & 31, wave = t >> 5;
  const int half = lane >> 4, rr = lane & 15, kh = half * 8;
  const size_t r0 = (size_t)blockIdx.x * 128 + wave * 16;   // (b,m) row tile

  // ---- Y[16,64] = Z[16,32] . pb[32,64]  (4 WMMA, K = p padded to 32)
  const bf16* za = &zb[(r0 + rr) * 32];
  v16bf a = cat16(ld8(&za[kh]), ld8(&za[kh + 16]));
  v8f y[4] = {};
#pragma unroll
  for (int jt = 0; jt < 4; ++jt) {
    v16bf bv;
#pragma unroll
    for (int s = 0; s < 16; ++s)
      bv[s] = pb[(kh + (s & 7) + (s >> 3) * 16) * 64 + jt * 16 + rr];
    y[jt] = wmma_bf16(a, bv, y[jt]);
  }
  // relayout Y (C layout) -> A fragments via LDS
#pragma unroll
  for (int jt = 0; jt < 4; ++jt)
#pragma unroll
    for (int v = 0; v < 8; ++v)
      Ys[wave][(v + 8 * half) * LDY + jt * 16 + rr] = (bf16)y[jt][v];
  __syncthreads();

  // ---- U[16,16] = Y[16,64] . wlb  (K=64 as 2x32 WMMA)
  v8f u = {};
#pragma unroll
  for (int kt = 0; kt < 2; ++kt) {
    const bf16* ab = &Ys[wave][rr * LDY + kt * 32 + kh];
    v16bf a2 = cat16(ld8(&ab[0]), ld8(&ab[16]));
    v16bf bv;
#pragma unroll
    for (int s = 0; s < 16; ++s)
      bv[s] = wlb[(kt * 32 + kh + (s & 7) + (s >> 3) * 16) * 16 + rr];
    u = wmma_bf16(a2, bv, u);
  }

  // bias + swish, then max over the 16 channel lanes (e = lane&15)
  const float bl = b_lin[rr];
  float val[8];
#pragma unroll
  for (int v = 0; v < 8; ++v) {
    float xx = u[v] + bl;
    val[v] = xx / (1.f + __expf(-xx));
  }
#pragma unroll
  for (int v = 0; v < 8; ++v)
#pragma unroll
    for (int m = 1; m < 16; m <<= 1)
      val[v] = fmaxf(val[v], __shfl_xor(val[v], m, 16));
  if (rr == 0) {
#pragma unroll
    for (int v = 0; v < 8; ++v)
      sout[r0 + v + 8 * half] = val[v];
  }
}

// ---------------------------------------------------------------- tiny head
__global__ __launch_bounds__(256) void head(const float* __restrict__ s,
                                            const float* __restrict__ W1,
                                            const float* __restrict__ b1,
                                            float* __restrict__ out) {
  __shared__ float red[256][10];
  const int t = threadIdx.x, b = blockIdx.x;
  float acc[10];
#pragma unroll
  for (int o = 0; o < 10; ++o) acc[o] = 0.f;
  for (int m = t; m < 784; m += 256) {
    float sv = s[(size_t)b * 784 + m];
#pragma unroll
    for (int o = 0; o < 10; ++o) acc[o] += sv * W1[o * 784 + m];
  }
#pragma unroll
  for (int o = 0; o < 10; ++o) red[t][o] = acc[o];
  __syncthreads();
  for (int st = 128; st > 0; st >>= 1) {
    if (t < st)
#pragma unroll
      for (int o = 0; o < 10; ++o) red[t][o] += red[t + st][o];
    __syncthreads();
  }
  if (t < 10) out[b * 10 + t] = red[0][t] + b1[t];
}

// ---------------------------------------------------------------- launcher
extern "C" void kernel_launch(void* const* d_in, const int* in_sizes, int n_in,
                              void* d_out, int out_size, void* d_ws, size_t ws_size,
                              hipStream_t stream) {
  const float* x      = (const float*)d_in[0];
  const float* T      = (const float*)d_in[1];
  const float* params = (const float*)d_in[2];
  const float* W_lin  = (const float*)d_in[3];
  const float* b_lin  = (const float*)d_in[4];
  const float* W1     = (const float*)d_in[5];
  const float* b1     = (const float*)d_in[6];
  float* out = (float*)d_out;

  char* ws = (char*)d_ws;
  bf16*  xb   = (bf16*)(ws);                      // 256*800*2      =   409600 B
  bf16*  pb   = (bf16*)(ws + 409600);             // 32*64*2        =     4096 B
  bf16*  wlb  = (bf16*)(ws + 413696);             // 64*16*2        =     2048 B
  bf16*  zb   = (bf16*)(ws + 415744);             // 200704*32*2    = 12845056 B
  float* sbuf = (float*)(ws + 13260800);          // 200704*4       =   802816 B
                                                  // total ~14.1 MB of d_ws

  prep_x    <<<800,          256, 0, stream>>>(x, xb);
  prep_small<<<784,          256, 0, stream>>>(params, W_lin, pb, wlb, zb);
  gemm1     <<<dim3(217, 1), 256, 0, stream>>>(xb, T, zb);
  stage2    <<<1568,         256, 0, stream>>>(zb, pb, wlb, b_lin, sbuf);
  head      <<<256,          256, 0, stream>>>(sbuf, W1, b1, out);
}
